// SelfAttentionBlock_88356067214056
// MI455X (gfx1250) — matrix operand order repaired and checked by reading, not compiled, not run on hardware
//
#include <hip/hip_runtime.h>
#include <hip/hip_bf16.h>

// ---------------------------------------------------------------------------
// SelfAttentionBlock for N=32, C=H*W=1024 on gfx1250 (MI455X).
// bf16 WMMA GEMM pipeline, f32 accumulation. All GEMM operands are laid out
// K-contiguous, so both A and B tiles stage via global_load_async_to_lds_b128
// (ASYNCcnt path), double-buffered LDS, one barrier per K-tile.
// The Q projection is algebraically eliminated: qsum = Wq * (sum_s nx) + S*bq.
// ---------------------------------------------------------------------------

typedef __attribute__((ext_vector_type(16))) __bf16        v16bf;
typedef __attribute__((ext_vector_type(8)))  float         v8f;
typedef __attribute__((ext_vector_type(4)))  unsigned int  u32x4;

#define NB    32
#define CDIM  1024
#define SDIM  1024
#define CS    (1024 * 1024)   // C * S elements per sample
#define KSTEP 64
#define KTILES (1024 / KSTEP)

union FragBF { v16bf v; u32x4 q[2]; };
union Pack4  { __bf16 h[4]; unsigned long long u; };

// --- CDNA5 async global->LDS copy (ASYNCcnt path) --------------------------
// Copies one 128B row slice (KSTEP bf16) with 8 b128 ops; INST_OFFSET applies
// to both the LDS and global addresses, so one address pair serves all chunks.
__device__ __forceinline__ void async_row128B(unsigned lds_addr, const void* gaddr) {
    asm volatile(
        "global_load_async_to_lds_b128 %0, %1, off\n"
        "global_load_async_to_lds_b128 %0, %1, off offset:16\n"
        "global_load_async_to_lds_b128 %0, %1, off offset:32\n"
        "global_load_async_to_lds_b128 %0, %1, off offset:48\n"
        "global_load_async_to_lds_b128 %0, %1, off offset:64\n"
        "global_load_async_to_lds_b128 %0, %1, off offset:80\n"
        "global_load_async_to_lds_b128 %0, %1, off offset:96\n"
        "global_load_async_to_lds_b128 %0, %1, off offset:112\n"
        :: "v"(lds_addr), "v"(gaddr) : "memory");
}
__device__ __forceinline__ void wait_async0() {
#if __has_builtin(__builtin_amdgcn_s_wait_asynccnt)
    __builtin_amdgcn_s_wait_asynccnt(0);
#else
    asm volatile("s_wait_asynccnt 0x0" ::: "memory");
#endif
}
__device__ __forceinline__ unsigned lds_off(const void* p) {
    return (unsigned)(unsigned long long)p;   // AS3 offset lives in low 32 bits
}

// ---------------------------------------------------------------------------
// 1) Per-sample mean / rstd over 1M elements
// ---------------------------------------------------------------------------
__global__ void stats_kernel(const float* __restrict__ x, float* __restrict__ stats) {
    __shared__ float s_sum[256];
    __shared__ float s_sq[256];
    const int b = blockIdx.x, tid = threadIdx.x;
    const float4* xb = (const float4*)(x + (size_t)b * CS);
    float sum = 0.f, sq = 0.f;
    for (int i = tid; i < CS / 4; i += 256) {
        float4 v = xb[i];
        sum += v.x + v.y + v.z + v.w;
        sq  += v.x * v.x + v.y * v.y + v.z * v.z + v.w * v.w;
    }
    s_sum[tid] = sum; s_sq[tid] = sq;
    __syncthreads();
    for (int o = 128; o > 0; o >>= 1) {
        if (tid < o) { s_sum[tid] += s_sum[tid + o]; s_sq[tid] += s_sq[tid + o]; }
        __syncthreads();
    }
    if (tid == 0) {
        float mu  = s_sum[0] * (1.0f / CS);
        float var = s_sq[0] * (1.0f / CS) - mu * mu;
        stats[b]      = mu;
        stats[NB + b] = rsqrtf(var + 1e-5f);
    }
}

// ---------------------------------------------------------------------------
// 2) LayerNorm -> bf16, written TRANSPOSED: nxT[b][s][c] (K-contiguous for
//    the projection GEMMs). 64x64 tile transpose through LDS per block.
// ---------------------------------------------------------------------------
__global__ void norm_t_kernel(const float* __restrict__ x, const float* __restrict__ lnw,
                              const float* __restrict__ lnb, const float* __restrict__ stats,
                              __bf16* __restrict__ nxT) {
    __shared__ __bf16 tile[64][66];           // [c_local][s_local], padded
    const int s0 = blockIdx.x * 64, c0 = blockIdx.y * 64, b = blockIdx.z;
    const int tid = threadIdx.x;
    const float mu = stats[b], rs = stats[NB + b];
    const int tr  = tid >> 4;                 // 0..15
    const int tc4 = (tid & 15) * 4;           // 0..60 step 4
    const float* xb = x + (size_t)b * CS;
#pragma unroll
    for (int p = 0; p < 4; p++) {
        const int cl = tr + p * 16;
        const size_t idx = (size_t)(c0 + cl) * 1024 + s0 + tc4;
        float4 xv = *(const float4*)(xb + idx);
        float4 wv = *(const float4*)(lnw + idx);
        float4 bv = *(const float4*)(lnb + idx);
        tile[cl][tc4 + 0] = (__bf16)((xv.x - mu) * rs * wv.x + bv.x);
        tile[cl][tc4 + 1] = (__bf16)((xv.y - mu) * rs * wv.y + bv.y);
        tile[cl][tc4 + 2] = (__bf16)((xv.z - mu) * rs * wv.z + bv.z);
        tile[cl][tc4 + 3] = (__bf16)((xv.w - mu) * rs * wv.w + bv.w);
    }
    __syncthreads();
    __bf16* ob = nxT + (size_t)b * CS;
#pragma unroll
    for (int p = 0; p < 4; p++) {
        const int sl = tr + p * 16;
        Pack4 t;
        t.h[0] = tile[tc4 + 0][sl];
        t.h[1] = tile[tc4 + 1][sl];
        t.h[2] = tile[tc4 + 2][sl];
        t.h[3] = tile[tc4 + 3][sl];
        *(unsigned long long*)(ob + (size_t)(s0 + sl) * 1024 + c0 + tc4) = t.u;
    }
}

// ---------------------------------------------------------------------------
// 3) fp32 -> bf16 weight conversion
// ---------------------------------------------------------------------------
__global__ void cvt_bf16_kernel(const float* __restrict__ in, __bf16* __restrict__ out, int n4) {
    int i = blockIdx.x * 256 + threadIdx.x;
    if (i < n4) {
        float4 v = ((const float4*)in)[i];
        Pack4 t;
        t.h[0] = (__bf16)v.x; t.h[1] = (__bf16)v.y;
        t.h[2] = (__bf16)v.z; t.h[3] = (__bf16)v.w;
        ((unsigned long long*)out)[i] = t.u;
    }
}

// ---------------------------------------------------------------------------
// 4) nxsum[b][c] = sum_s nxT[b][s][c]   (column reduction, coalesced in c)
// ---------------------------------------------------------------------------
__global__ void nxsum_kernel(const __bf16* __restrict__ nxT, float* __restrict__ nxsum) {
    const int c = blockIdx.x * 256 + threadIdx.x;
    const int b = blockIdx.y;
    const __bf16* base = nxT + (size_t)b * CS + c;
    float s = 0.f;
#pragma unroll 4
    for (int srow = 0; srow < SDIM; srow++) s += (float)base[(size_t)srow * 1024];
    nxsum[b * 1024 + c] = s;
}

// ---------------------------------------------------------------------------
// 5) qsum[b][o] = dot(wq[o][:], nxsum[b][:]) + 1024 * bq[o]
//    (Q projection eliminated: only its spatial sum is ever needed.)
// ---------------------------------------------------------------------------
__global__ void qsum_gemv_kernel(const float* __restrict__ wq, const float* __restrict__ bq,
                                 const float* __restrict__ nxsum, float* __restrict__ qsum) {
    __shared__ float red[256];
    const int o = blockIdx.x, b = blockIdx.y, tid = threadIdx.x;
    const float* wrow = wq + (size_t)o * 1024;
    const float* ns   = nxsum + b * 1024;
    float s = 0.f;
    for (int i = tid; i < 1024; i += 256) s += wrow[i] * ns[i];
    red[tid] = s; __syncthreads();
    for (int off = 128; off > 0; off >>= 1) {
        if (tid < off) red[tid] += red[tid + off];
        __syncthreads();
    }
    if (tid == 0) qsum[b * 1024 + o] = red[0] + 1024.0f * bq[o];
}

// ---------------------------------------------------------------------------
// 6) Batched bf16 GEMM, M=N=K=1024:
//      D[m][n] = sum_k A[m][k] * Bt[n][k]   (both operands K-contiguous)
//    128x128 block tile, 4 waves, 64x64 per wave = 4x4 WMMA 16x16x32 bf16.
//    K-step 64; A and B tiles both staged by async global->LDS copies,
//    double-buffered, one barrier per K-tile.
// ---------------------------------------------------------------------------
__global__ __launch_bounds__(128) void gemm_bf16_kernel(
    const __bf16* __restrict__ A,  long long sA,   // [M][K]
    const __bf16* __restrict__ Bt, long long sB,   // [N][K]
    const float*  __restrict__ bias,               // per-output-row, or null
    const float*  __restrict__ resid, long long sR,
    void* __restrict__ out, long long sO,
    int out_bf16, int tstore)
{
    __shared__ __align__(16) unsigned short ldsA[2][128 * KSTEP];  // [row][k]
    __shared__ __align__(16) unsigned short ldsB[2][128 * KSTEP];  // [col][k]

    const int tid  = threadIdx.x;
    const int lane = tid & 31;
    const int wave = tid >> 5;
    const int wm   = wave >> 1;          // 2x2 wave grid over 128x128
    const int wn   = wave & 1;
    const int hs   = lane >> 4;          // half-select (lanes 16..31)
    const int l15  = lane & 15;

    const int bn = blockIdx.x * 128;
    const int bm = blockIdx.y * 128;
    const int b  = blockIdx.z;

    const __bf16* aRow = A  + (size_t)b * (size_t)sA + (size_t)(bm + tid) * 1024;
    const __bf16* bRow = Bt + (size_t)b * (size_t)sB + (size_t)(bn + tid) * 1024;

    const unsigned aL[2] = { lds_off(&ldsA[0][tid * KSTEP]), lds_off(&ldsA[1][tid * KSTEP]) };
    const unsigned bL[2] = { lds_off(&ldsB[0][tid * KSTEP]), lds_off(&ldsB[1][tid * KSTEP]) };

    v8f zero = {0.f, 0.f, 0.f, 0.f, 0.f, 0.f, 0.f, 0.f};
    v8f acc[4][4];
#pragma unroll
    for (int i = 0; i < 4; i++)
#pragma unroll
        for (int j = 0; j < 4; j++) acc[i][j] = zero;

    // ---- prologue: stage tile 0
    async_row128B(aL[0], aRow);
    async_row128B(bL[0], bRow);

    for (int it = 0; it < KTILES; ++it) {
        const int cur = it & 1;

        wait_async0();       // this wave's tile-it copies have landed in LDS
        __syncthreads();     // (compiler also drains pending ds loads here)

        // prefetch next K-tile into the other buffer
        if (it + 1 < KTILES) {
            const int kk = (it + 1) * KSTEP;
            async_row128B(aL[cur ^ 1], aRow + kk);
            async_row128B(bL[cur ^ 1], bRow + kk);
        }

        // ---- compute: 2 sub-steps x 16 WMMA from current buffers
        const unsigned short* la = &ldsA[cur][0];
        const unsigned short* lb = &ldsB[cur][0];
#pragma unroll
        for (int ks = 0; ks < KSTEP; ks += 32) {
            FragBF af[4], bfr[4];
#pragma unroll
            for (int i = 0; i < 4; i++) {
                int row = wm * 64 + i * 16 + l15;
                const unsigned short* p = la + row * KSTEP + ks + hs * 8;
                af[i].q[0] = *(const u32x4*)p;
                af[i].q[1] = *(const u32x4*)(p + 16);
            }
#pragma unroll
            for (int j = 0; j < 4; j++) {
                int col = wn * 64 + j * 16 + l15;
                const unsigned short* p = lb + col * KSTEP + ks + hs * 8;
                bfr[j].q[0] = *(const u32x4*)p;
                bfr[j].q[1] = *(const u32x4*)(p + 16);
            }
#pragma unroll
            for (int i = 0; i < 4; i++)
#pragma unroll
                for (int j = 0; j < 4; j++)
                    acc[i][j] = __builtin_amdgcn_wmma_f32_16x16x32_bf16(
                        false, af[i].v, false, bfr[j].v, (short)0, acc[i][j], false, false);
        }
    }

    // ---- epilogue
#pragma unroll
    for (int i = 0; i < 4; i++) {
        const int gm = bm + wm * 64 + i * 16;
#pragma unroll
        for (int j = 0; j < 4; j++) {
            const int gn  = bn + wn * 64 + j * 16;
            const int col = gn + l15;
#pragma unroll
            for (int r = 0; r < 8; r++) {
                const int row = gm + hs * 8 + r;   // ISA C/D layout: VGPR r -> M
                float v = acc[i][j][r];
                if (bias)  v += bias[row];
                if (resid) v += resid[(size_t)b * (size_t)sR + (size_t)row * 1024 + col];
                size_t oi = (size_t)b * (size_t)sO +
                            (tstore ? ((size_t)col * 1024 + row)
                                    : ((size_t)row * 1024 + col));
                if (out_bf16) ((__bf16*)out)[oi] = (__bf16)v;
                else          ((float*)out)[oi]  = v;
            }
        }
    }
}

// ---------------------------------------------------------------------------
// 7) p[b][j][c] = softmax_c( qsum[b][c] * kT[b][j][c] / sqrt(C) )
//    kT is the transposed K projection -> fully coalesced reads along c.
// ---------------------------------------------------------------------------
__global__ void softmax_kernel(const __bf16* __restrict__ kT, const float* __restrict__ qsum,
                               __bf16* __restrict__ p) {
    __shared__ float red[256];
    const int j = blockIdx.x, b = blockIdx.y, tid = threadIdx.x;
    const __bf16* kj = kT + (size_t)b * CS + (size_t)j * 1024;
    const float*  qs = qsum + b * 1024;
    float vals[4];
    float m = -1e30f;
#pragma unroll
    for (int i = 0; i < 4; i++) {
        int c = tid + i * 256;
        float v = (float)kj[c] * qs[c] * 0.03125f;   // 1/sqrt(1024)
        vals[i] = v;
        m = fmaxf(m, v);
    }
    red[tid] = m; __syncthreads();
    for (int o = 128; o > 0; o >>= 1) {
        if (tid < o) red[tid] = fmaxf(red[tid], red[tid + o]);
        __syncthreads();
    }
    m = red[0]; __syncthreads();
    float s = 0.f;
#pragma unroll
    for (int i = 0; i < 4; i++) { vals[i] = __expf(vals[i] - m); s += vals[i]; }
    red[tid] = s; __syncthreads();
    for (int o = 128; o > 0; o >>= 1) {
        if (tid < o) red[tid] += red[tid + o];
        __syncthreads();
    }
    const float inv = 1.0f / red[0];
    __bf16* pj = p + (size_t)b * CS + (size_t)j * 1024;
#pragma unroll
    for (int i = 0; i < 4; i++) pj[tid + i * 256] = (__bf16)(vals[i] * inv);
}

// ---------------------------------------------------------------------------
// Launcher
// ---------------------------------------------------------------------------
extern "C" void kernel_launch(void* const* d_in, const int* in_sizes, int n_in,
                              void* d_out, int out_size, void* d_ws, size_t ws_size,
                              hipStream_t stream) {
    (void)in_sizes; (void)n_in; (void)out_size; (void)ws_size;

    const float* x   = (const float*)d_in[0];
    const float* lnw = (const float*)d_in[1];
    const float* lnb = (const float*)d_in[2];
    const float* wq  = (const float*)d_in[3];
    const float* bq  = (const float*)d_in[4];
    const float* wk  = (const float*)d_in[5];
    const float* bk  = (const float*)d_in[6];
    const float* wv  = (const float*)d_in[7];
    const float* bv  = (const float*)d_in[8];
    const float* wo  = (const float*)d_in[9];
    const float* bo  = (const float*)d_in[10];

    char* ws = (char*)d_ws;
    const size_t MB = 1024ull * 1024ull;
    float*  stats = (float*)ws;                                   // 256 B
    __bf16* nx    = (__bf16*)(ws + 4096);                          // 64 MB; nxT, reused as probs p
    __bf16* wkb   = (__bf16*)(ws + 4096 + 64 * MB);                // bf16 weights (K, V, O)
    __bf16* wvb   = wkb + (size_t)CDIM * CDIM;
    __bf16* wob   = wvb + (size_t)CDIM * CDIM;
    __bf16* ab    = (__bf16*)(ws + 4096 + 72 * MB);                // 64 MB; attn
    __bf16* kb    = ab + (size_t)NB * CS;                          // 64 MB; kT
    __bf16* vb    = kb + (size_t)NB * CS;                          // 64 MB; vT
    float*  qsum  = (float*)(ws + 4096 + 72 * MB + 192 * MB);      // 128 KB
    float*  nxsum = qsum + (size_t)NB * CDIM;                      // 128 KB

    // 1) LayerNorm statistics
    stats_kernel<<<NB, 256, 0, stream>>>(x, stats);
    // 2) normalize + transpose -> nxT[b][s][c] (bf16)
    norm_t_kernel<<<dim3(SDIM / 64, CDIM / 64, NB), 256, 0, stream>>>(x, lnw, lnb, stats, nx);
    // 3) weights -> bf16 (wq not needed in bf16: Q GEMM is eliminated)
    cvt_bf16_kernel<<<1024, 256, 0, stream>>>(wk, wkb, CDIM * CDIM / 4);
    cvt_bf16_kernel<<<1024, 256, 0, stream>>>(wv, wvb, CDIM * CDIM / 4);
    cvt_bf16_kernel<<<1024, 256, 0, stream>>>(wo, wob, CDIM * CDIM / 4);

    // 4) qsum via linearity: nxsum = sum_s nxT; qsum = Wq*nxsum + 1024*bq
    nxsum_kernel<<<dim3(CDIM / 256, NB), 256, 0, stream>>>(nx, nxsum);
    qsum_gemv_kernel<<<dim3(CDIM, NB), 256, 0, stream>>>(wq, bq, nxsum, qsum);

    dim3 gg(SDIM / 128, CDIM / 128, NB);
    // 5) K/V projections with transposed store: kT[s][c], vT[s][c]
    gemm_bf16_kernel<<<gg, 128, 0, stream>>>(wkb, 0, nx, CS, bk, nullptr, 0, kb, CS, 1, 1);
    gemm_bf16_kernel<<<gg, 128, 0, stream>>>(wvb, 0, nx, CS, bv, nullptr, 0, vb, CS, 1, 1);

    // 6) softmax (probs overwrite nxT buffer, which is now dead)
    softmax_kernel<<<dim3(SDIM, NB), 256, 0, stream>>>(kb, qsum, nx);

    // 7) attn[j][s] = sum_c p[j][c] * vT[s][c]  (normal store)
    gemm_bf16_kernel<<<gg, 128, 0, stream>>>(nx, CS, vb, CS, nullptr, nullptr, 0, ab, CS, 1, 0);

    // 8) out[o][j] = sum_s wo[o][s] * attn[j][s] + bo[o] + x  (f32 -> d_out)
    gemm_bf16_kernel<<<gg, 128, 0, stream>>>(wob, 0, ab, CS, bo, x, CS, d_out, CS, 0, 0);
}
